// MultiheadAttention_51161650430216
// MI455X (gfx1250) — matrix operand order repaired and checked by reading, not compiled, not run on hardware
//
#include <hip/hip_runtime.h>
#include <hip/hip_bf16.h>
#include <math.h>

#define D_MODEL 1024
#define NUM_HEADS 16
#define HEAD_DIM 64
#define SEQ 2048
#define BATCH 2
#define ROWS (BATCH * SEQ)   // 4096 token rows

typedef __attribute__((ext_vector_type(16))) __bf16 bf16x16;
typedef __attribute__((ext_vector_type(8)))  float  v8f;

union Frag { uint4 u[2]; bf16x16 v; };

// ---------------------------------------------------------------------------
// WMMA wrapper: D = A(16x32 bf16) * B(32x16 bf16) + C(16x16 f32)
// ---------------------------------------------------------------------------
__device__ __forceinline__ v8f wmma_bf16(bf16x16 a, bf16x16 b, v8f c) {
  return __builtin_amdgcn_wmma_f32_16x16x32_bf16(
      /*neg_a=*/false, a, /*neg_b=*/false, b,
      /*c_mod=*/(short)0, c, /*reuse_a=*/false, /*reuse_b=*/false);
}

// A fragment (16x32, 16-bit): lane l holds row M = l&15,
// K = k0 + (l>>4)*8 + {0..7} in v0..3 and +16 in v4..7  -> two 16B loads.
__device__ __forceinline__ bf16x16 load_frag_a(const __bf16* base, int row0,
                                               int ld, int k0, int lane) {
  const __bf16* p = base + (size_t)(row0 + (lane & 15)) * ld + k0 + ((lane >> 4) << 3);
  Frag f;
  f.u[0] = *(const uint4*)(p);
  f.u[1] = *(const uint4*)(p + 16);
  return f.v;
}

// B fragment (32x16, 16-bit) where B = matᵀ, i.e. B[k][n] = mat[n][k] and
// mat is row-major with leading dim ld.  Lane l holds column N = n0 + (l&15),
// K = k0 + (l>>4)*8 + {0..7} in v0..3 and +16 in v4..7  -> two 16B loads.
__device__ __forceinline__ bf16x16 load_frag_bt(const __bf16* mat, int n0,
                                                int ld, int k0, int lane) {
  const __bf16* p = mat + (size_t)(n0 + (lane & 15)) * ld + k0 + ((lane >> 4) << 3);
  Frag f;
  f.u[0] = *(const uint4*)(p);
  f.u[1] = *(const uint4*)(p + 16);
  return f.v;
}

// ---------------------------------------------------------------------------
// fp32 -> bf16 conversion (4 elements / thread)
// ---------------------------------------------------------------------------
__global__ __launch_bounds__(256) void cvt_f32_bf16(const float* __restrict__ src,
                                                    __bf16* __restrict__ dst, int n) {
  int i = (blockIdx.x * blockDim.x + threadIdx.x) * 4;
  if (i < n) {
    float4 f = *(const float4*)(src + i);
    union { __bf16 b[4]; uint2 u; } o;
    o.b[0] = (__bf16)f.x; o.b[1] = (__bf16)f.y;
    o.b[2] = (__bf16)f.z; o.b[3] = (__bf16)f.w;
    *(uint2*)(dst + i) = o.u;
  }
}

// ---------------------------------------------------------------------------
// GEMM: out = A[ROWS x 1024] @ W[1024 x 1024]^T
//   mode 0: store bf16 into [B,H,S,Dh]   (Q, K)
//   mode 1: store bf16 into [B,H,Dh,S]   (V transposed for the PV B-fragment)
//   mode 2: store fp32 into [ROWS x 1024] (final output)
// Block = 256 threads = 8 waves arranged 4(row) x 2(col).
// Block tile 128x128; wave tile 32x64 (2 A-frags x 4 B-frags = 8 WMMA/k-step).
// Software-pipelined: next k-step fragments loaded before current WMMAs issue.
// ---------------------------------------------------------------------------
__global__ __launch_bounds__(256) void gemm_bf16(const __bf16* __restrict__ A,
                                                 const __bf16* __restrict__ W,
                                                 void* __restrict__ outp, int mode) {
  const int lane    = threadIdx.x & 31;
  const int wave    = threadIdx.x >> 5;
  const int rowbase = blockIdx.y * 128 + (wave >> 1) * 32;
  const int colbase = blockIdx.x * 128 + (wave & 1) * 64;

  v8f acc[2][4] = {};

  // prologue: k-step 0
  bf16x16 a0 = load_frag_a(A, rowbase,      D_MODEL, 0, lane);
  bf16x16 a1 = load_frag_a(A, rowbase + 16, D_MODEL, 0, lane);
  bf16x16 b0 = load_frag_bt(W, colbase,      D_MODEL, 0, lane);
  bf16x16 b1 = load_frag_bt(W, colbase + 16, D_MODEL, 0, lane);
  bf16x16 b2 = load_frag_bt(W, colbase + 32, D_MODEL, 0, lane);
  bf16x16 b3 = load_frag_bt(W, colbase + 48, D_MODEL, 0, lane);

  for (int k0 = 0; k0 < D_MODEL; k0 += 32) {
    // rotate current fragments into dedicated regs
    bf16x16 ca0 = a0, ca1 = a1;
    bf16x16 cb0 = b0, cb1 = b1, cb2 = b2, cb3 = b3;

    // prefetch next k-step (clamped on the last iteration: redundant but safe)
    const int kn = (k0 + 32 < D_MODEL) ? (k0 + 32) : k0;
    a0 = load_frag_a(A, rowbase,      D_MODEL, kn, lane);
    a1 = load_frag_a(A, rowbase + 16, D_MODEL, kn, lane);
    b0 = load_frag_bt(W, colbase,      D_MODEL, kn, lane);
    b1 = load_frag_bt(W, colbase + 16, D_MODEL, kn, lane);
    b2 = load_frag_bt(W, colbase + 32, D_MODEL, kn, lane);
    b3 = load_frag_bt(W, colbase + 48, D_MODEL, kn, lane);

    acc[0][0] = wmma_bf16(ca0, cb0, acc[0][0]);
    acc[1][0] = wmma_bf16(ca1, cb0, acc[1][0]);
    acc[0][1] = wmma_bf16(ca0, cb1, acc[0][1]);
    acc[1][1] = wmma_bf16(ca1, cb1, acc[1][1]);
    acc[0][2] = wmma_bf16(ca0, cb2, acc[0][2]);
    acc[1][2] = wmma_bf16(ca1, cb2, acc[1][2]);
    acc[0][3] = wmma_bf16(ca0, cb3, acc[0][3]);
    acc[1][3] = wmma_bf16(ca1, cb3, acc[1][3]);
  }

  // C/D layout: element (m,n): m = r + 8*(lane>>4), n = lane&15
  const int nloc = lane & 15;
#pragma unroll
  for (int rt = 0; rt < 2; ++rt) {
    const int mbase = rowbase + rt * 16 + ((lane >> 4) << 3);
#pragma unroll
    for (int t = 0; t < 4; ++t) {
#pragma unroll
      for (int r = 0; r < 8; ++r) {
        int row = mbase + r;
        int col = colbase + t * 16 + nloc;
        float val = acc[rt][t][r];
        if (mode == 2) {
          ((float*)outp)[(size_t)row * D_MODEL + col] = val;
        } else {
          int b  = row >> 11, s = row & (SEQ - 1);
          int h  = col >> 6,  dh = col & (HEAD_DIM - 1);
          size_t idx = (mode == 0)
              ? ((size_t)((b * NUM_HEADS + h) * SEQ + s) * HEAD_DIM + dh)
              : ((size_t)((b * NUM_HEADS + h) * HEAD_DIM + dh) * SEQ + s);
          ((__bf16*)outp)[idx] = (__bf16)val;
        }
      }
    }
  }
}

// ---------------------------------------------------------------------------
// Streaming attention (no max-subtraction softmax, faithful to reference):
//   num += exp(QKᵀ/8 + mask) @ V ; den += rowsum(exp) ; out = num/(den+EPS)
// One wave owns a 16-row query tile; streams keys in steps of 32.
// All global loads of an iteration are hoisted to the top so they overlap the
// score WMMAs, the expf VALU work and the LDS P round-trip.
// ---------------------------------------------------------------------------
__global__ __launch_bounds__(256) void attn_kernel(const __bf16* __restrict__ Q,
                                                   const __bf16* __restrict__ K,
                                                   const __bf16* __restrict__ VT,
                                                   const float* __restrict__ mask,
                                                   __bf16* __restrict__ AO) {
  // per-wave P staging tile (16 x 32 bf16, padded rows to stagger LDS banks)
  __shared__ __align__(16) __bf16 plds[8][16][40];

  const int lane  = threadIdx.x & 31;
  const int wave  = threadIdx.x >> 5;
  const int bh    = blockIdx.y;              // b*16 + h
  const int b     = bh >> 4;
  const int h     = bh & 15;
  const int qbase = blockIdx.x * 128 + wave * 16;   // query row within sequence

  const __bf16* q  = Q  + (size_t)bh * SEQ * HEAD_DIM;
  const __bf16* kk = K  + (size_t)bh * SEQ * HEAD_DIM;
  const __bf16* vt = VT + (size_t)bh * HEAD_DIM * SEQ;

  // Q fragments covering Dh=64 (two 16x32 A-frags), loaded once
  bf16x16 qa0 = load_frag_a(q, qbase, HEAD_DIM, 0,  lane);
  bf16x16 qa1 = load_frag_a(q, qbase, HEAD_DIM, 32, lane);

  v8f   oacc[4] = {};
  float den[8]  = {0.f, 0.f, 0.f, 0.f, 0.f, 0.f, 0.f, 0.f};

  const int mloc = (lane >> 4) << 3;   // 0 or 8
  const int nloc = lane & 15;
  const float* mrow = mask + (size_t)(qbase + mloc) * SEQ + nloc;

  for (int t0 = 0; t0 < SEQ; t0 += 32) {
    // speculative prefetch of the next key tile (global_prefetch_b8)
    __builtin_prefetch(kk + (size_t)(t0 + 32) * HEAD_DIM, 0, 1);

    // ---- hoist ALL global loads of this iteration -------------------------
    bf16x16 kb00 = load_frag_bt(kk, t0,      HEAD_DIM, 0,  lane);
    bf16x16 kb01 = load_frag_bt(kk, t0,      HEAD_DIM, 32, lane);
    bf16x16 kb10 = load_frag_bt(kk, t0 + 16, HEAD_DIM, 0,  lane);
    bf16x16 kb11 = load_frag_bt(kk, t0 + 16, HEAD_DIM, 32, lane);
    bf16x16 vb0  = load_frag_bt(vt, 0,  SEQ, t0, lane);
    bf16x16 vb1  = load_frag_bt(vt, 16, SEQ, t0, lane);
    bf16x16 vb2  = load_frag_bt(vt, 32, SEQ, t0, lane);
    bf16x16 vb3  = load_frag_bt(vt, 48, SEQ, t0, lane);
    float mv[2][8];
#pragma unroll
    for (int sub = 0; sub < 2; ++sub)
#pragma unroll
      for (int r = 0; r < 8; ++r)
        mv[sub][r] = mrow[(size_t)r * SEQ + t0 + sub * 16];

    // ---- scores for the two 16-column subtiles ----------------------------
    v8f s0 = {}, s1 = {};
    s0 = wmma_bf16(qa0, kb00, s0);
    s0 = wmma_bf16(qa1, kb01, s0);
    s1 = wmma_bf16(qa0, kb10, s1);
    s1 = wmma_bf16(qa1, kb11, s1);

    // ---- exp, denominator accumulation, stage P in LDS --------------------
#pragma unroll
    for (int r = 0; r < 8; ++r) {
      float p0 = expf(s0[r] * 0.125f + mv[0][r]);
      float p1 = expf(s1[r] * 0.125f + mv[1][r]);
      den[r] += p0 + p1;
      plds[wave][mloc + r][nloc]      = (__bf16)p0;
      plds[wave][mloc + r][16 + nloc] = (__bf16)p1;
    }

    // ---- reload P as an A-fragment (16x32) from LDS -----------------------
    const __bf16* prow = &plds[wave][lane & 15][(lane >> 4) << 3];
    Frag pf;
    pf.u[0] = *(const uint4*)(prow);
    pf.u[1] = *(const uint4*)(prow + 16);
    bf16x16 pa = pf.v;

    // ---- P @ V : B[t][dh] = VT[dh][t], dh-columns contiguous over t -------
    oacc[0] = wmma_bf16(pa, vb0, oacc[0]);
    oacc[1] = wmma_bf16(pa, vb1, oacc[1]);
    oacc[2] = wmma_bf16(pa, vb2, oacc[2]);
    oacc[3] = wmma_bf16(pa, vb3, oacc[3]);
  }

  // ---- reduce row sums across the 16 lanes of each N-group ----------------
#pragma unroll
  for (int r = 0; r < 8; ++r) {
    float d = den[r];
    d += __shfl_xor(d, 1);
    d += __shfl_xor(d, 2);
    d += __shfl_xor(d, 4);
    d += __shfl_xor(d, 8);
    den[r] = d + 1e-10f;
  }

  // ---- normalize and store attn-out as bf16 [B, S, H*Dh] ------------------
#pragma unroll
  for (int d4 = 0; d4 < 4; ++d4) {
#pragma unroll
    for (int r = 0; r < 8; ++r) {
      int row = qbase + mloc + r;
      int col = h * HEAD_DIM + d4 * 16 + nloc;
      AO[(size_t)(b * SEQ + row) * D_MODEL + col] = (__bf16)(oacc[d4][r] / den[r]);
    }
  }
}

// ---------------------------------------------------------------------------
// Host-side orchestration
// ---------------------------------------------------------------------------
extern "C" void kernel_launch(void* const* d_in, const int* in_sizes, int n_in,
                              void* d_out, int out_size, void* d_ws, size_t ws_size,
                              hipStream_t stream) {
  const float* x    = (const float*)d_in[0];
  const float* mask = (const float*)d_in[1];
  const float* Wq   = (const float*)d_in[2];
  const float* Wk   = (const float*)d_in[3];
  const float* Wv   = (const float*)d_in[4];
  const float* Wo   = (const float*)d_in[5];

  char* ws = (char*)d_ws;
  const size_t xbytes = (size_t)ROWS * D_MODEL * sizeof(__bf16);      // 8 MiB
  const size_t wbytes = (size_t)D_MODEL * D_MODEL * sizeof(__bf16);   // 2 MiB
  const size_t hbytes = (size_t)BATCH * NUM_HEADS * SEQ * HEAD_DIM * sizeof(__bf16); // 8 MiB

  __bf16* xb  = (__bf16*)ws;             ws += xbytes;
  __bf16* wqb = (__bf16*)ws;             ws += wbytes;
  __bf16* wkb = (__bf16*)ws;             ws += wbytes;
  __bf16* wvb = (__bf16*)ws;             ws += wbytes;
  __bf16* wob = (__bf16*)ws;             ws += wbytes;
  __bf16* qb  = (__bf16*)ws;             ws += hbytes;   // [B,H,S,Dh]
  __bf16* kb  = (__bf16*)ws;             ws += hbytes;   // [B,H,S,Dh]
  __bf16* vtb = (__bf16*)ws;             ws += hbytes;   // [B,H,Dh,S]
  __bf16* aob = (__bf16*)ws;             ws += xbytes;   // [B,S,D]

  const int nx = ROWS * D_MODEL;
  const int nw = D_MODEL * D_MODEL;
  cvt_f32_bf16<<<nx / 4 / 256, 256, 0, stream>>>(x,  xb,  nx);
  cvt_f32_bf16<<<nw / 4 / 256, 256, 0, stream>>>(Wq, wqb, nw);
  cvt_f32_bf16<<<nw / 4 / 256, 256, 0, stream>>>(Wk, wkb, nw);
  cvt_f32_bf16<<<nw / 4 / 256, 256, 0, stream>>>(Wv, wvb, nw);
  cvt_f32_bf16<<<nw / 4 / 256, 256, 0, stream>>>(Wo, wob, nw);

  dim3 gg(D_MODEL / 128, ROWS / 128);    // (8, 32)
  gemm_bf16<<<gg, 256, 0, stream>>>(xb, wqb, qb,  0);
  gemm_bf16<<<gg, 256, 0, stream>>>(xb, wkb, kb,  0);
  gemm_bf16<<<gg, 256, 0, stream>>>(xb, wvb, vtb, 1);

  dim3 ga(SEQ / 128, BATCH * NUM_HEADS); // (16, 32)
  attn_kernel<<<ga, 256, 0, stream>>>(qb, kb, vtb, mask, aob);

  gemm_bf16<<<gg, 256, 0, stream>>>(aob, wob, d_out, 2);
}
